// AdaLNSelfAttn_56839597195378
// MI455X (gfx1250) — matrix operand order repaired and checked by reading, not compiled, not run on hardware
//
#include <hip/hip_runtime.h>
#include <hip/hip_bf16.h>
#include <math.h>

// ---------------------------------------------------------------------------
// AdaLN self-attention block for MI455X (gfx1250, wave32, WMMA bf16).
//   B=2, L=2048, C=1024, H=16, DH=64, DFF=4096.
// fp32 -> bf16 once, all GEMMs + attention on v_wmma_f32_16x16x32_bf16 with
// f32 accumulation. Working set (~134MB) fits the 192MB L2, so GEMMs read
// operands directly from global (L2-resident weights / attn_bias).
// GEMM K-loops are software-pipelined (double-buffered fragments) so the
// s_wait_loadcnt before each WMMA covers loads issued one iteration earlier.
// ---------------------------------------------------------------------------

#define C_   1024
#define H_   16
#define DH_  64
#define DFF_ 4096
#define D_   1024
#define B_   2
#define L_   2048
#define MROW (B_ * L_)            // 4096 token rows

typedef __attribute__((ext_vector_type(16))) __bf16 v16bf;
typedef __attribute__((ext_vector_type(8)))  __bf16 v8bf;
typedef __attribute__((ext_vector_type(8)))  float  v8f;

__device__ __forceinline__ unsigned short f2bf(float f) {
  union { float f; unsigned u; } v; v.f = f;
  unsigned r = v.u + 0x7FFFu + ((v.u >> 16) & 1u);   // round-to-nearest-even
  return (unsigned short)(r >> 16);
}

__device__ __forceinline__ float gelu_tanh(float v) {
  float u = 0.7978845608028654f * (v + 0.044715f * v * v * v);
  return 0.5f * v * (1.0f + tanhf(u));
}

#define WMMA_BF16(a, b, c) \
  __builtin_amdgcn_wmma_f32_16x16x32_bf16(false, (a), false, (b), (short)0, (c), false, false)

// ---------------------------------------------------------------------------
// Wave-level GEMM core: one wave computes a 16(M) x 64(N) f32 tile of
// A[M,K](bf16, row-major) @ W[N,K](bf16, row-major)^T, K multiple of 32.
// A-frag layout (16x32 bf16 A): lane<16 -> K {0..15}; lane>=16 ->
// K {16..31}. B-frag: lane holds 16 consecutive K for column n=lane%16
// (contiguous in W row n). C/D: lane j, vgpr r -> (m = r + 8*(j>=16), n = j%16).
// Double-buffered: next iteration's 6 fragment loads are issued before the
// current 4 WMMAs consume their operands.
// ---------------------------------------------------------------------------
__device__ __forceinline__ void wave_gemm_16x64(
    const unsigned short* __restrict__ A, const unsigned short* __restrict__ W,
    int K, int rowA, int rowW, v8f acc[4]) {
  const int lane = threadIdx.x & 31;
  const int hi = lane >> 4;          // half-wave group
  const int l16 = lane & 15;
  const unsigned short* ap = A + (size_t)(rowA + l16) * K + hi * 16;
  const unsigned short* wp = W + (size_t)(rowW + l16) * K + hi * 16;
  const size_t ws16 = (size_t)16 * K;

  // prologue: fragments for kk = 0
  v8bf  a0 = *(const v8bf*)(ap);
  v8bf  a1 = *(const v8bf*)(ap + 8);
  v16bf b0 = *(const v16bf*)(wp);
  v16bf b1 = *(const v16bf*)(wp + ws16);
  v16bf b2 = *(const v16bf*)(wp + 2 * ws16);
  v16bf b3 = *(const v16bf*)(wp + 3 * ws16);

  for (int kk = 0; kk < K; kk += 32) {
    v8bf  na0 = a0, na1 = a1;
    v16bf nb0 = b0, nb1 = b1, nb2 = b2, nb3 = b3;
    if (kk + 32 < K) {   // issue next-iteration loads before consuming current
      const unsigned short* apn = ap + kk + 32;
      const unsigned short* wpn = wp + kk + 32;
      __builtin_prefetch(apn + 128, 0, 1);          // global_prefetch_b8
      na0 = *(const v8bf*)(apn);
      na1 = *(const v8bf*)(apn + 8);
      nb0 = *(const v16bf*)(wpn);
      nb1 = *(const v16bf*)(wpn + ws16);
      nb2 = *(const v16bf*)(wpn + 2 * ws16);
      nb3 = *(const v16bf*)(wpn + 3 * ws16);
    }
    v16bf a;
#pragma unroll
    for (int i = 0; i < 8; i++) { a[i] = a0[i]; a[i + 8] = a1[i]; }
    acc[0] = WMMA_BF16(a, b0, acc[0]);
    acc[1] = WMMA_BF16(a, b1, acc[1]);
    acc[2] = WMMA_BF16(a, b2, acc[2]);
    acc[3] = WMMA_BF16(a, b3, acc[3]);
    a0 = na0; a1 = na1; b0 = nb0; b1 = nb1; b2 = nb2; b3 = nb3;
  }
}

// wave/tile decode shared by GEMM kernels (8 waves: 4 in M x 2 in N)
#define GEMM_PROLOG()                                                    \
  const int w = threadIdx.x >> 5;                                        \
  const int tileM = blockIdx.y * 64 + (w & 3) * 16;                      \
  const int tileN = blockIdx.x * 128 + (w >> 2) * 64;                    \
  const int lane = threadIdx.x & 31;                                     \
  const int hi = lane >> 4;                                              \
  const int l16 = lane & 15;                                             \
  v8f acc[4] = {};                                                       \
  (void)lane;

// --- qkv = hA @ qkv_w^T + bias3 -> fp32 [MROW, 3C] ---
__global__ void __launch_bounds__(256)
gemm_qkv_kernel(const unsigned short* __restrict__ A,
                const unsigned short* __restrict__ W,
                const float* __restrict__ bias3, float* __restrict__ out) {
  GEMM_PROLOG()
  wave_gemm_16x64(A, W, C_, tileM, tileN, acc);
#pragma unroll
  for (int t = 0; t < 4; t++) {
    int gn = tileN + t * 16 + l16;
    float bv = bias3[gn];
#pragma unroll
    for (int r = 0; r < 8; r++) {
      int gm = tileM + hi * 8 + r;
      out[(size_t)gm * (3 * C_) + gn] = acc[t][r] + bv;
    }
  }
}

// --- x1 = x + (attn_o @ proj_w^T + proj_b) * g1 -> fp32 [MROW, C] ---
__global__ void __launch_bounds__(256)
gemm_proj_kernel(const unsigned short* __restrict__ A,
                 const unsigned short* __restrict__ W,
                 const float* __restrict__ pb, const float* __restrict__ x,
                 const float* __restrict__ ada, float* __restrict__ x1) {
  GEMM_PROLOG()
  wave_gemm_16x64(A, W, C_, tileM, tileN, acc);
#pragma unroll
  for (int t = 0; t < 4; t++) {
    int gn = tileN + t * 16 + l16;
    float bv = pb[gn];
#pragma unroll
    for (int r = 0; r < 8; r++) {
      int gm = tileM + hi * 8 + r;
      int b = gm >> 11;  // row / L
      float g1 = ada[((size_t)b * 6 + 0) * C_ + gn];
      x1[(size_t)gm * C_ + gn] =
          x[(size_t)gm * C_ + gn] + (acc[t][r] + bv) * g1;
    }
  }
}

// --- ff1 = gelu(h2 @ fc1_w^T + fc1_b) -> bf16 [MROW, DFF] ---
__global__ void __launch_bounds__(256)
gemm_fc1_kernel(const unsigned short* __restrict__ A,
                const unsigned short* __restrict__ W,
                const float* __restrict__ fb, unsigned short* __restrict__ out) {
  GEMM_PROLOG()
  wave_gemm_16x64(A, W, C_, tileM, tileN, acc);
#pragma unroll
  for (int t = 0; t < 4; t++) {
    int gn = tileN + t * 16 + l16;
    float bv = fb[gn];
#pragma unroll
    for (int r = 0; r < 8; r++) {
      int gm = tileM + hi * 8 + r;
      out[(size_t)gm * DFF_ + gn] = f2bf(gelu_tanh(acc[t][r] + bv));
    }
  }
}

// --- out = x1 + (ff1 @ fc2_w^T + fc2_b) * g2 -> fp32 [MROW, C] ---
__global__ void __launch_bounds__(256)
gemm_fc2_kernel(const unsigned short* __restrict__ A,
                const unsigned short* __restrict__ W,
                const float* __restrict__ fb, const float* __restrict__ x1,
                const float* __restrict__ ada, float* __restrict__ out) {
  GEMM_PROLOG()
  wave_gemm_16x64(A, W, DFF_, tileM, tileN, acc);
#pragma unroll
  for (int t = 0; t < 4; t++) {
    int gn = tileN + t * 16 + l16;
    float bv = fb[gn];
#pragma unroll
    for (int r = 0; r < 8; r++) {
      int gm = tileM + hi * 8 + r;
      int b = gm >> 11;
      float g2 = ada[((size_t)b * 6 + 1) * C_ + gn];
      out[(size_t)gm * C_ + gn] =
          x1[(size_t)gm * C_ + gn] + (acc[t][r] + bv) * g2;
    }
  }
}

// ---------------------------------------------------------------------------
// Flash attention: one wave per 16 query rows, online softmax over 32-key
// tiles. Row stats live at vgpr slot r for row (r + 8*hi) and are reduced
// across the 16-lane half-group with shfl_xor(1/2/4/8) (never crosses bit 4).
// P is re-swizzled C-layout -> A-layout through a per-wave LDS tile.
// All global loads of an iteration (K frags, V frags, bias) are issued up
// front so they are in flight during the softmax VALU work.
// ---------------------------------------------------------------------------
__global__ void __launch_bounds__(256)
attn_flash_kernel(const unsigned short* __restrict__ qn,
                  const unsigned short* __restrict__ kn,
                  const unsigned short* __restrict__ vT,
                  const float* __restrict__ bias,
                  unsigned short* __restrict__ attn_o) {
  __shared__ __align__(16) unsigned short pl[8][16][32];  // 8KB
  const int w = threadIdx.x >> 5;
  const int lane = threadIdx.x & 31;
  const int hi = lane >> 4;
  const int l16 = lane & 15;
  const int b = blockIdx.z, h = blockIdx.y;
  const int q0 = blockIdx.x * 128 + w * 16;  // query base within L
  const size_t bh = (size_t)(b * H_ + h);
  const unsigned short* qbase = qn + (bh * L_ + q0) * DH_;
  const unsigned short* kbase = kn + bh * L_ * DH_ + (size_t)l16 * DH_ + hi * 16;
  const unsigned short* vbase = vT + bh * DH_ * L_ + (size_t)l16 * L_ + hi * 16;
  const float* brow = bias + (size_t)(q0 + hi * 8) * L_ + l16;

  // Q A-frags: DH=64 -> two 16x32 bf16 fragments, resident for whole loop
  v16bf aq0, aq1;
  {
    const unsigned short* qr = qbase + (size_t)l16 * DH_;
    v8bf t0 = *(const v8bf*)(qr + hi * 16);
    v8bf t1 = *(const v8bf*)(qr + 8 + hi * 16);
    v8bf t2 = *(const v8bf*)(qr + 32 + hi * 16);
    v8bf t3 = *(const v8bf*)(qr + 40 + hi * 16);
#pragma unroll
    for (int i = 0; i < 8; i++) {
      aq0[i] = t0[i]; aq0[i + 8] = t1[i];
      aq1[i] = t2[i]; aq1[i + 8] = t3[i];
    }
  }

  v8f o[4] = {};
  float mrun[8], lrun[8];
#pragma unroll
  for (int r = 0; r < 8; r++) { mrun[r] = -3.0e38f; lrun[r] = 0.0f; }

  for (int k0 = 0; k0 < L_; k0 += 32) {
    // ---- issue ALL global loads for this iteration up front ----
    const unsigned short* kr0 = kbase + (size_t)k0 * DH_;            // keys k0..k0+15
    const unsigned short* kr1 = kbase + (size_t)(k0 + 16) * DH_;     // keys k0+16..k0+31
    v16bf kb00 = *(const v16bf*)(kr0);
    v16bf kb01 = *(const v16bf*)(kr0 + 32);
    v16bf kb10 = *(const v16bf*)(kr1);
    v16bf kb11 = *(const v16bf*)(kr1 + 32);
    v16bf bv[4];                                  // V frags (independent of softmax)
#pragma unroll
    for (int t = 0; t < 4; t++)
      bv[t] = *(const v16bf*)(vbase + (size_t)(t * 16) * L_ + k0);
    float bl0[8], bl1[8];                         // attn_bias (shared over b,h)
#pragma unroll
    for (int r = 0; r < 8; r++) {
      bl0[r] = brow[(size_t)r * L_ + k0];
      bl1[r] = brow[(size_t)r * L_ + k0 + 16];
    }

    // ---- S = Q @ K^T (two 16-key n-tiles, K-dim = DH via 2 WMMAs each) ----
    v8f s0 = {}, s1 = {};
    s0 = WMMA_BF16(aq0, kb00, s0);
    s0 = WMMA_BF16(aq1, kb01, s0);
    s1 = WMMA_BF16(aq0, kb10, s1);
    s1 = WMMA_BF16(aq1, kb11, s1);
#pragma unroll
    for (int r = 0; r < 8; r++) { s0[r] = s0[r] + bl0[r]; s1[r] = s1[r] + bl1[r]; }

    // ---- online softmax ----
    float p0a[8], p1a[8], cf[8];
#pragma unroll
    for (int r = 0; r < 8; r++) {
      float v = fmaxf(s0[r], s1[r]);
      v = fmaxf(v, __shfl_xor(v, 1, 32));
      v = fmaxf(v, __shfl_xor(v, 2, 32));
      v = fmaxf(v, __shfl_xor(v, 4, 32));
      v = fmaxf(v, __shfl_xor(v, 8, 32));
      float mnew = fmaxf(mrun[r], v);
      cf[r] = __expf(mrun[r] - mnew);
      mrun[r] = mnew;
      p0a[r] = __expf(s0[r] - mnew);
      p1a[r] = __expf(s1[r] - mnew);
      float rs = p0a[r] + p1a[r];
      rs += __shfl_xor(rs, 1, 32);
      rs += __shfl_xor(rs, 2, 32);
      rs += __shfl_xor(rs, 4, 32);
      rs += __shfl_xor(rs, 8, 32);
      lrun[r] = lrun[r] * cf[r] + rs;
    }
#pragma unroll
    for (int t = 0; t < 4; t++)
#pragma unroll
      for (int r = 0; r < 8; r++) o[t][r] = o[t][r] * cf[r];

    // ---- P: C-layout regs -> LDS [16 q][32 k] -> A-layout frag ----
    __syncthreads();
#pragma unroll
    for (int r = 0; r < 8; r++) {
      pl[w][hi * 8 + r][l16] = f2bf(p0a[r]);
      pl[w][hi * 8 + r][16 + l16] = f2bf(p1a[r]);
    }
    __syncthreads();
    v16bf ap;
    {
      v8bf t0 = *(const v8bf*)&pl[w][l16][hi * 16];
      v8bf t1 = *(const v8bf*)&pl[w][l16][8 + hi * 16];
#pragma unroll
      for (int i = 0; i < 8; i++) { ap[i] = t0[i]; ap[i + 8] = t1[i]; }
    }
    // ---- O += P @ V (K-dim = 32 keys; V pre-transposed [dh][l]) ----
#pragma unroll
    for (int t = 0; t < 4; t++) o[t] = WMMA_BF16(ap, bv[t], o[t]);
  }

  // normalize + write [B,L,H*DH] bf16
#pragma unroll
  for (int t = 0; t < 4; t++)
#pragma unroll
    for (int r = 0; r < 8; r++) {
      int qrow = q0 + hi * 8 + r;
      float val = o[t][r] / lrun[r];
      attn_o[((size_t)(b * L_ + qrow)) * C_ + h * DH_ + t * 16 + l16] = f2bf(val);
    }
}

// ---------------------------------------------------------------------------
// Elementwise / reduction helper kernels
// ---------------------------------------------------------------------------
__global__ void silu_kernel(const float* __restrict__ c, float* __restrict__ o, int n) {
  int i = blockIdx.x * 256 + threadIdx.x;
  if (i < n) { float v = c[i]; o[i] = v / (1.0f + __expf(-v)); }
}

__global__ void ada_gemm_kernel(const float* __restrict__ sc,
                                const float* __restrict__ ada_w,
                                const float* __restrict__ ada_b,
                                float* __restrict__ ada) {
  int o = blockIdx.x * 256 + threadIdx.x;   // 0 .. 6C-1
  int b = blockIdx.y;
  const float* wr = ada_w + (size_t)o * D_;
  const float* cv = sc + (size_t)b * D_;
  float s = 0.0f;
  for (int k = 0; k < D_; k++) s += cv[k] * wr[k];
  ada[(size_t)b * (6 * C_) + o] = s + ada_b[o];
}

__global__ void bias3_kernel(const float* __restrict__ qb,
                             const float* __restrict__ vb,
                             float* __restrict__ b3) {
  int i = blockIdx.x * 256 + threadIdx.x;   // 0 .. 3C-1
  float v = (i < C_) ? qb[i] : ((i < 2 * C_) ? 0.0f : vb[i - 2 * C_]);
  b3[i] = v;
}

__global__ void cvt_bf16_kernel(const float* __restrict__ in,
                                unsigned short* __restrict__ out, int n) {
  int i = blockIdx.x * 256 + threadIdx.x;
  if (i < n) out[i] = f2bf(in[i]);
}

// LayerNorm (no affine, eps 1e-6) + adaLN modulate -> bf16. One block per row.
__global__ void __launch_bounds__(256)
ln_mod_kernel(const float* __restrict__ x, const float* __restrict__ ada,
              int cs, int csh, unsigned short* __restrict__ out) {
  const int row = blockIdx.x;
  const int b = row >> 11;
  const int tid = threadIdx.x;
  __shared__ float red[256];
  __shared__ float s_mean, s_rstd;
  float4 xv = ((const float4*)(x + (size_t)row * C_))[tid];
  red[tid] = xv.x + xv.y + xv.z + xv.w;
  __syncthreads();
  for (int st = 128; st > 0; st >>= 1) {
    if (tid < st) red[tid] += red[tid + st];
    __syncthreads();
  }
  if (tid == 0) s_mean = red[0] * (1.0f / C_);
  __syncthreads();
  float m = s_mean;
  float d0 = xv.x - m, d1 = xv.y - m, d2 = xv.z - m, d3 = xv.w - m;
  __syncthreads();
  red[tid] = d0 * d0 + d1 * d1 + d2 * d2 + d3 * d3;
  __syncthreads();
  for (int st = 128; st > 0; st >>= 1) {
    if (tid < st) red[tid] += red[tid + st];
    __syncthreads();
  }
  if (tid == 0) s_rstd = rsqrtf(red[0] * (1.0f / C_) + 1e-6f);
  __syncthreads();
  float rstd = s_rstd;
  const float* sa = ada + ((size_t)b * 6 + cs) * C_ + tid * 4;
  const float* sha = ada + ((size_t)b * 6 + csh) * C_ + tid * 4;
  unsigned short* orow = out + (size_t)row * C_ + tid * 4;
  float dv[4] = {d0, d1, d2, d3};
#pragma unroll
  for (int i = 0; i < 4; i++)
    orow[i] = f2bf(dv[i] * rstd * (sa[i] + 1.0f) + sha[i]);
}

// qkv post: per-(b,h,l) L2-normalize q (scaled by exp(min(scale_mul,ln100)))
// and k over DH; pack q,k bf16 head-major; pack v transposed [b,h,dh,l].
__global__ void __launch_bounds__(256)
qkv_post_kernel(const float* __restrict__ qkv, const float* __restrict__ scale_mul,
                unsigned short* __restrict__ qn, unsigned short* __restrict__ kn,
                unsigned short* __restrict__ vT) {
  const int wave = (blockIdx.x << 3) + (threadIdx.x >> 5);
  const int lane = threadIdx.x & 31;
  const int l = wave & (L_ - 1);
  const int h = (wave >> 11) & (H_ - 1);
  const int b = wave >> 15;
  const float* base = qkv + ((size_t)b * L_ + l) * (3 * C_) + h * DH_;
  float sm = __expf(fminf(scale_mul[h], 4.6051702f));  // log(100)

  float q0 = base[lane], q1 = base[lane + 32];
  float sq = q0 * q0 + q1 * q1;
#pragma unroll
  for (int off = 16; off; off >>= 1) sq += __shfl_xor(sq, off, 32);
  float rq = rsqrtf(fmaxf(sq, 1e-24f)) * sm;

  float k0 = base[C_ + lane], k1 = base[C_ + lane + 32];
  float sk = k0 * k0 + k1 * k1;
#pragma unroll
  for (int off = 16; off; off >>= 1) sk += __shfl_xor(sk, off, 32);
  float rk = rsqrtf(fmaxf(sk, 1e-24f));

  const size_t ob = ((size_t)(b * H_ + h) * L_ + l) * DH_;
  qn[ob + lane] = f2bf(q0 * rq);
  qn[ob + lane + 32] = f2bf(q1 * rq);
  kn[ob + lane] = f2bf(k0 * rk);
  kn[ob + lane + 32] = f2bf(k1 * rk);

  float v0 = base[2 * C_ + lane], v1 = base[2 * C_ + lane + 32];
  const size_t vb0 = ((size_t)(b * H_ + h) * DH_) * L_ + l;
  vT[vb0 + (size_t)lane * L_] = f2bf(v0);
  vT[vb0 + (size_t)(lane + 32) * L_] = f2bf(v1);
}

// ---------------------------------------------------------------------------
extern "C" void kernel_launch(void* const* d_in, const int* in_sizes, int n_in,
                              void* d_out, int out_size, void* d_ws, size_t ws_size,
                              hipStream_t stream) {
  (void)in_sizes; (void)n_in; (void)out_size; (void)ws_size;
  const float* x         = (const float*)d_in[0];
  const float* cond      = (const float*)d_in[1];
  const float* attn_bias = (const float*)d_in[2];
  const float* qkv_w     = (const float*)d_in[3];
  const float* q_bias    = (const float*)d_in[4];
  const float* v_bias    = (const float*)d_in[5];
  const float* scale_mul = (const float*)d_in[6];
  const float* proj_w    = (const float*)d_in[7];
  const float* proj_b    = (const float*)d_in[8];
  const float* fc1_w     = (const float*)d_in[9];
  const float* fc1_b     = (const float*)d_in[10];
  const float* fc2_w     = (const float*)d_in[11];
  const float* fc2_b     = (const float*)d_in[12];
  const float* ada_w     = (const float*)d_in[13];
  const float* ada_b     = (const float*)d_in[14];
  float* out = (float*)d_out;

  // Workspace bump allocator (~134 MB total; buffers reused where safe).
  char* ws = (char*)d_ws;
  size_t off = 0;
  auto alloc = [&](size_t bytes) -> char* {
    char* p = ws + off;
    off += (bytes + 255) & ~(size_t)255;
    return p;
  };
  float* silu_c = (float*)alloc((size_t)B_ * D_ * 4);
  float* ada    = (float*)alloc((size_t)B_ * 6 * C_ * 4);
  float* bias3  = (float*)alloc((size_t)3 * C_ * 4);
  unsigned short* wq = (unsigned short*)alloc((size_t)3 * C_ * C_ * 2);
  unsigned short* wp = (unsigned short*)alloc((size_t)C_ * C_ * 2);
  unsigned short* w1 = (unsigned short*)alloc((size_t)DFF_ * C_ * 2);
  unsigned short* w2 = (unsigned short*)alloc((size_t)C_ * DFF_ * 2);
  unsigned short* hA = (unsigned short*)alloc((size_t)MROW * C_ * 2);   // LN1 out; reused for LN2 out
  float* qkvbuf = (float*)alloc((size_t)MROW * 3 * C_ * 4);             // reused as ff1 (bf16) later
  unsigned short* qn = (unsigned short*)alloc((size_t)B_ * H_ * L_ * DH_ * 2);
  unsigned short* kn = (unsigned short*)alloc((size_t)B_ * H_ * L_ * DH_ * 2);
  unsigned short* vT = (unsigned short*)alloc((size_t)B_ * H_ * L_ * DH_ * 2);
  unsigned short* attn_o = (unsigned short*)alloc((size_t)MROW * C_ * 2);
  float* x1 = (float*)alloc((size_t)MROW * C_ * 4);
  unsigned short* ff1 = (unsigned short*)qkvbuf;

  // 1) adaLN modulation
  silu_kernel<<<(B_ * D_ + 255) / 256, 256, 0, stream>>>(cond, silu_c, B_ * D_);
  ada_gemm_kernel<<<dim3(6 * C_ / 256, B_), 256, 0, stream>>>(silu_c, ada_w, ada_b, ada);
  bias3_kernel<<<3 * C_ / 256, 256, 0, stream>>>(q_bias, v_bias, bias3);

  // 2) weights fp32 -> bf16 (once per launch; deterministic)
  cvt_bf16_kernel<<<(3 * C_ * C_) / 256, 256, 0, stream>>>(qkv_w, wq, 3 * C_ * C_);
  cvt_bf16_kernel<<<(C_ * C_) / 256, 256, 0, stream>>>(proj_w, wp, C_ * C_);
  cvt_bf16_kernel<<<(DFF_ * C_) / 256, 256, 0, stream>>>(fc1_w, w1, DFF_ * C_);
  cvt_bf16_kernel<<<(C_ * DFF_) / 256, 256, 0, stream>>>(fc2_w, w2, C_ * DFF_);

  // 3) attention branch
  ln_mod_kernel<<<MROW, 256, 0, stream>>>(x, ada, 2, 4, hA);             // s1, sh1
  gemm_qkv_kernel<<<dim3(3 * C_ / 128, MROW / 64), 256, 0, stream>>>(hA, wq, bias3, qkvbuf);
  qkv_post_kernel<<<(B_ * H_ * L_) / 8, 256, 0, stream>>>(qkvbuf, scale_mul, qn, kn, vT);
  attn_flash_kernel<<<dim3(L_ / 128, H_, B_), 256, 0, stream>>>(qn, kn, vT, attn_bias, attn_o);
  gemm_proj_kernel<<<dim3(C_ / 128, MROW / 64), 256, 0, stream>>>(attn_o, wp, proj_b, x, ada, x1);

  // 4) FFN branch
  ln_mod_kernel<<<MROW, 256, 0, stream>>>(x1, ada, 3, 5, hA);            // s2, sh2
  gemm_fc1_kernel<<<dim3(DFF_ / 128, MROW / 64), 256, 0, stream>>>(hA, w1, fc1_b, ff1);
  gemm_fc2_kernel<<<dim3(C_ / 128, MROW / 64), 256, 0, stream>>>(ff1, w2, fc2_b, x1, ada, out);
}
